// DeeperGCN_87565793231060
// MI455X (gfx1250) — matrix operand order repaired and checked by reading, not compiled, not run on hardware
//
#include <hip/hip_runtime.h>
#include <cstdint>
#include <cstddef>

// ---- problem constants (match reference) ----
#define NN    20000     // nodes
#define NE    320000    // edges
#define FIN   128
#define HD    256       // hidden
#define HD2   512       // 2*hidden
#define NGR   64        // graphs
#define NOUT  64
#define EPS_MSG 1e-7f
#define LN_EPS  1e-5f

typedef float    v2f  __attribute__((ext_vector_type(2)));
typedef float    v8f  __attribute__((ext_vector_type(8)));
typedef _Float16 v16h __attribute__((ext_vector_type(16)));

#if __has_builtin(__builtin_amdgcn_wmma_f32_16x16x4_f32)
#define HAVE_WMMA_F32X4 1
#endif

#define GF_RELU 1
#define GF_ACC  2

// ---------------------------------------------------------------------------
// WMMA GEMM:  C[M,N] (+)= A[M,K] @ W[K,N] + bias[N]   (row-major, all f32)
// Block = 256 threads = 8 waves. Block tile = 16 rows x 128 cols
// (one 16x16 WMMA tile per wave). A-tile (16xK) staged in LDS; B panel
// prefetched (global_prefetch_b8) while staging. M%16==0, N%128==0, K%32==0
// for all call sites here.
// ---------------------------------------------------------------------------
__global__ __launch_bounds__(256)
void k_gemm(const float* __restrict__ A, const float* __restrict__ W,
            const float* __restrict__ bias, float* __restrict__ C,
            int M, int N, int K, int flags)
{
    __shared__ __align__(16) float As[16 * 512];   // K <= 512
    const int tid  = threadIdx.x;
    const int lane = tid & 31;
    const int wid  = tid >> 5;
    const int row0 = blockIdx.y << 4;
    const int col0 = (blockIdx.x << 7) + (wid << 4);

    // warm the B panel (one prefetch per K-row of the 128-col panel)
    for (int k = tid; k < K; k += 256)
        __builtin_prefetch(W + (size_t)k * N + (blockIdx.x << 7), 0, 0);

    // ---- stage A tile [16 x K] into LDS (float4 granularity) ----
    const int kv4 = K >> 2;
    for (int idx = tid; idx < (kv4 << 4); idx += 256) {
        const int r  = idx / kv4;
        const int kv = idx - r * kv4;
        const float* gp = A + (size_t)(row0 + r) * K + (kv << 2);
        float* lp = &As[r * K + (kv << 2)];
        *(float4*)lp = *(const float4*)gp;
    }
    __syncthreads();

    if (col0 < N) {
        const int m  = lane & 15;        // row within A tile / col within B
        const int hi = lane >> 4;        // lane half
        const int cn = col0 + m;         // output column for this lane
        const float* arow = &As[m * K];
        const float* bp   = W + cn;
        v8f acc = {};
#ifdef HAVE_WMMA_F32X4
        const int kh = hi << 1;          // K offset for this lane half (f32 x4)
        for (int k0 = 0; k0 < K; k0 += 4) {
            v2f a, b;
            a[0] = arow[k0 + kh];
            a[1] = arow[k0 + kh + 1];
            b[0] = bp[(size_t)(k0 + kh)     * N];
            b[1] = bp[(size_t)(k0 + kh + 1) * N];
            acc = __builtin_amdgcn_wmma_f32_16x16x4_f32(
                false, a, false, b, (short)0, acc, false, false);
        }
#else
        // fallback: probe-confirmed f16 WMMA (K step 32), on-the-fly convert
        for (int k0 = 0; k0 < K; k0 += 32) {
            v16h a, b;
            #pragma unroll
            for (int j = 0; j < 16; ++j) {
                const int ka = k0 + j + ((j >> 3) << 3) + (hi << 3);
                const int kb = k0 + j + ((j >> 3) << 3) + (hi << 3);
                a[j] = (_Float16)arow[ka];
                b[j] = (_Float16)bp[(size_t)kb * N];
            }
            acc = __builtin_amdgcn_wmma_f32_16x16x32_f16(
                false, a, false, b, (short)0, acc, false, false);
        }
#endif
        // C/D layout: VGPR r -> row (r + 8*hi), col = lane&15
        const float bv    = bias[cn];
        const int   rbase = row0 + (hi << 3);
        #pragma unroll
        for (int r = 0; r < 8; ++r) {
            float v = acc[r] + bv;
            if (flags & GF_RELU) v = fmaxf(v, 0.f);
            float* cp = &C[(size_t)(rbase + r) * N + cn];
            if (flags & GF_ACC) *cp += v; else *cp = v;
        }
    }
}

// ---------------------------------------------------------------------------
// Edge phase: segment softmax in 3 passes (block-per-edge, thread-per-feature)
// ---------------------------------------------------------------------------
__device__ __forceinline__ int f2ord(float x) {
    int i = __float_as_int(x);
    return (i >= 0) ? i : (i ^ 0x7FFFFFFF);      // order-preserving map
}
__device__ __forceinline__ float ord2f(int i) {
    int j = (i >= 0) ? i : (i ^ 0x7FFFFFFF);
    return __int_as_float(j);
}

__global__ __launch_bounds__(256)
void k_init_conv(int* __restrict__ mbuf, float* __restrict__ sbuf,
                 float* __restrict__ agg, int n)
{
    const int stride = gridDim.x * 256;
    for (int i = blockIdx.x * 256 + threadIdx.x; i < n; i += stride) {
        mbuf[i] = (int)0x80000000;   // INT_MIN == encoded -inf identity
        sbuf[i] = 0.f;
        agg[i]  = 0.f;
    }
}

__global__ __launch_bounds__(256)
void k_edge_max(const float* __restrict__ x, const int* __restrict__ src,
                const int* __restrict__ dst, int* __restrict__ mbuf,
                const float* __restrict__ tptr)
{
    const int e = blockIdx.x, f = threadIdx.x;
    const int s = src[e], d = dst[e];
    const float t   = *tptr;
    const float msg = fmaxf(x[(size_t)s * HD + f], 0.f) + EPS_MSG;
    atomicMax(&mbuf[(size_t)d * HD + f], f2ord(msg * t));
}

__global__ __launch_bounds__(256)
void k_edge_sum(const float* __restrict__ x, const int* __restrict__ src,
                const int* __restrict__ dst, const int* __restrict__ mbuf,
                float* __restrict__ sbuf, const float* __restrict__ tptr)
{
    const int e = blockIdx.x, f = threadIdx.x;
    const int s = src[e], d = dst[e];
    const float t   = *tptr;
    const float msg = fmaxf(x[(size_t)s * HD + f], 0.f) + EPS_MSG;
    const float m   = ord2f(mbuf[(size_t)d * HD + f]);
    atomicAdd(&sbuf[(size_t)d * HD + f], __expf(msg * t - m));
}

__global__ __launch_bounds__(256)
void k_edge_agg(const float* __restrict__ x, const int* __restrict__ src,
                const int* __restrict__ dst, const int* __restrict__ mbuf,
                const float* __restrict__ sbuf, float* __restrict__ agg,
                const float* __restrict__ tptr)
{
    const int e = blockIdx.x, f = threadIdx.x;
    const int s = src[e], d = dst[e];
    const float t     = *tptr;
    const float msg   = fmaxf(x[(size_t)s * HD + f], 0.f) + EPS_MSG;
    const size_t di   = (size_t)d * HD + f;
    const float ev    = __expf(msg * t - ord2f(mbuf[di]));
    const float alpha = ev / (sbuf[di] + 1e-16f);
    atomicAdd(&agg[di], msg * alpha);
}

__global__ __launch_bounds__(256)
void k_addroot(float* __restrict__ agg, const float* __restrict__ x, int n)
{
    const int stride = gridDim.x * 256;
    for (int i = blockIdx.x * 256 + threadIdx.x; i < n; i += stride)
        agg[i] += x[i];
}

// ---------------------------------------------------------------------------
// Row LayerNorm over width 512 + affine + ReLU, in place (block per row)
// ---------------------------------------------------------------------------
__global__ __launch_bounds__(256)
void k_ln_relu(float* __restrict__ h, const float* __restrict__ g,
               const float* __restrict__ b)
{
    __shared__ float s1[256], s2[256];
    const int row = blockIdx.x, tid = threadIdx.x;
    float* hp = h + (size_t)row * HD2;
    const float x0 = hp[tid], x1 = hp[tid + 256];
    s1[tid] = x0 + x1;
    s2[tid] = x0 * x0 + x1 * x1;
    __syncthreads();
    for (int s = 128; s > 0; s >>= 1) {
        if (tid < s) { s1[tid] += s1[tid + s]; s2[tid] += s2[tid + s]; }
        __syncthreads();
    }
    const float mu  = s1[0] * (1.f / HD2);
    const float var = s2[0] * (1.f / HD2) - mu * mu;
    const float inv = rsqrtf(var + LN_EPS);
    hp[tid]       = fmaxf((x0 - mu) * inv * g[tid]       + b[tid],       0.f);
    hp[tid + 256] = fmaxf((x1 - mu) * inv * g[tid + 256] + b[tid + 256], 0.f);
}

// ---------------------------------------------------------------------------
// Graph LayerNorm (over ALL nodes+features): reduce, then normalize
// ---------------------------------------------------------------------------
__global__ __launch_bounds__(256)
void k_zero(float* __restrict__ p, int n)
{
    const int stride = gridDim.x * 256;
    for (int i = blockIdx.x * 256 + threadIdx.x; i < n; i += stride) p[i] = 0.f;
}

__global__ __launch_bounds__(256)
void k_reduce(const float* __restrict__ x, float* __restrict__ red, int n)
{
    __shared__ float s1[256], s2[256];
    float a = 0.f, bq = 0.f;
    const int stride = gridDim.x * 256;
    for (int i = blockIdx.x * 256 + threadIdx.x; i < n; i += stride) {
        const float v = x[i];
        a += v; bq += v * v;
    }
    s1[threadIdx.x] = a; s2[threadIdx.x] = bq;
    __syncthreads();
    for (int s = 128; s > 0; s >>= 1) {
        if (threadIdx.x < s) { s1[threadIdx.x] += s1[threadIdx.x + s];
                               s2[threadIdx.x] += s2[threadIdx.x + s]; }
        __syncthreads();
    }
    if (threadIdx.x == 0) {
        atomicAdd(&red[0], s1[0]);
        atomicAdd(&red[1], s2[0]);
    }
}

__global__ __launch_bounds__(256)
void k_gln_relu(const float* __restrict__ x, float* __restrict__ out,
                const float* __restrict__ red, const float* __restrict__ g,
                const float* __restrict__ b, int n)
{
    const float invn  = 1.f / (float)n;
    const float mu    = red[0] * invn;
    const float var   = red[1] * invn - mu * mu;
    const float sd    = sqrtf(fmaxf(var, 0.f));
    const float scale = 1.f / (sd + LN_EPS);
    const int stride  = gridDim.x * 256;
    for (int i = blockIdx.x * 256 + threadIdx.x; i < n; i += stride) {
        const int c = i & (HD - 1);
        out[i] = fmaxf((x[i] - mu) * scale * g[c] + b[c], 0.f);
    }
}

// ---------------------------------------------------------------------------
// global_add_pool + tiny MLP head
// ---------------------------------------------------------------------------
__global__ __launch_bounds__(256)
void k_pool(const float* __restrict__ h, const int* __restrict__ batch,
            float* __restrict__ pool)
{
    const int nidx = blockIdx.x, f = threadIdx.x;
    const int gidx = batch[nidx];
    atomicAdd(&pool[(size_t)gidx * HD + f], h[(size_t)nidx * HD + f]);
}

__global__ __launch_bounds__(256)
void k_mlp1(const float* __restrict__ pool, const float* __restrict__ w,
            const float* __restrict__ b, float* __restrict__ tmp)
{
    const int idx = blockIdx.x * 256 + threadIdx.x;     // 64*128
    if (idx >= NGR * (HD / 2)) return;
    const int i = idx >> 7, j = idx & 127;
    float acc = b[j];
    for (int k = 0; k < HD; ++k) acc += pool[i * HD + k] * w[k * (HD / 2) + j];
    tmp[idx] = fmaxf(acc, 0.f);
}

__global__ __launch_bounds__(256)
void k_mlp2(const float* __restrict__ tmp, const float* __restrict__ w,
            const float* __restrict__ b, float* __restrict__ out)
{
    const int idx = blockIdx.x * 256 + threadIdx.x;     // 64*64
    if (idx >= NGR * NOUT) return;
    const int i = idx >> 6, j = idx & 63;
    float acc = b[j];
    for (int k = 0; k < HD / 2; ++k) acc += tmp[i * (HD / 2) + k] * w[k * NOUT + j];
    out[idx] = acc;
}

// ---------------------------------------------------------------------------
extern "C" void kernel_launch(void* const* d_in, const int* in_sizes, int n_in,
                              void* d_out, int out_size, void* d_ws, size_t ws_size,
                              hipStream_t stream)
{
    (void)in_sizes; (void)n_in; (void)out_size; (void)ws_size;

    const float* x     = (const float*)d_in[0];
    const int*   ei    = (const int*)d_in[1];
    const int*   batch = (const int*)d_in[2];
    const float* enc_w = (const float*)d_in[3];
    const float* enc_b = (const float*)d_in[4];
    // per-layer params: 0 w1, 1 b1, 2 lng, 3 lnb, 4 w2, 5 b2, 6 t, 7 ng, 8 nb
    const float* L[2][9];
    for (int l = 0; l < 2; ++l)
        for (int k = 0; k < 9; ++k)
            L[l][k] = (const float*)d_in[5 + l * 9 + k];
    const float* mlp_w1 = (const float*)d_in[23];
    const float* mlp_b1 = (const float*)d_in[24];
    const float* mlp_w2 = (const float*)d_in[25];
    const float* mlp_b2 = (const float*)d_in[26];
    const int* srcv = ei;
    const int* dstv = ei + NE;

    // ---- workspace layout (floats); seg-max/seg-sum alias the MID buffer ----
    float* ws   = (float*)d_ws;
    float* H0   = ws;                           // [NN,HD]
    float* H1   = H0  + (size_t)NN * HD;        // [NN,HD]
    float* Rb   = H1  + (size_t)NN * HD;        // [NN,HD]
    float* AGG  = Rb  + (size_t)NN * HD;        // [NN,HD]
    float* MID  = AGG + (size_t)NN * HD;        // [NN,HD2]  (aliased below)
    int*   MB   = (int*)MID;                    //   seg-max  [NN,HD]
    float* SB   = MID + (size_t)NN * HD;        //   seg-sum  [NN,HD]
    float* POOL = MID + (size_t)NN * HD2;       // [NGR,HD]
    float* TMP  = POOL + NGR * HD;              // [NGR,HD/2]
    float* RED  = TMP  + NGR * (HD / 2);        // [2]

    const dim3 B(256);

    // node encoder: H0 = x @ enc_w + enc_b
    k_gemm<<<dim3(HD / 128, NN / 16), B, 0, stream>>>(x, enc_w, enc_b, H0,
                                                      NN, HD, FIN, 0);

    auto conv = [&](const float* xin, const float* const* P, float* outC, int acc) {
        k_init_conv<<<4096, B, 0, stream>>>(MB, SB, AGG, NN * HD);
        k_edge_max<<<NE, B, 0, stream>>>(xin, srcv, dstv, MB, P[6]);
        k_edge_sum<<<NE, B, 0, stream>>>(xin, srcv, dstv, MB, SB, P[6]);
        k_edge_agg<<<NE, B, 0, stream>>>(xin, srcv, dstv, MB, SB, AGG, P[6]);
        k_addroot<<<4096, B, 0, stream>>>(AGG, xin, NN * HD);
        k_gemm<<<dim3(HD2 / 128, NN / 16), B, 0, stream>>>(AGG, P[0], P[1], MID,
                                                           NN, HD2, HD, 0);
        k_ln_relu<<<NN, B, 0, stream>>>(MID, P[2], P[3]);
        k_gemm<<<dim3(HD / 128, NN / 16), B, 0, stream>>>(MID, P[4], P[5], outC,
                                                          NN, HD, HD2, acc);
    };

    // layers[0].conv (bare): H1 = genconv(H0)
    conv(H0, L[0], H1, 0);

    // r = relu(graph_layernorm(H1, l1_ng, l1_nb))
    k_zero<<<1, B, 0, stream>>>(RED, 2);
    k_reduce<<<256, B, 0, stream>>>(H1, RED, NN * HD);
    k_gln_relu<<<2048, B, 0, stream>>>(H1, Rb, RED, L[1][7], L[1][8], NN * HD);

    // res+ layer: H1 += genconv(r)
    conv(Rb, L[1], H1, GF_ACC);

    // h = relu(graph_layernorm(H1, l0_ng, l0_nb)), in place
    k_zero<<<1, B, 0, stream>>>(RED, 2);
    k_reduce<<<256, B, 0, stream>>>(H1, RED, NN * HD);
    k_gln_relu<<<2048, B, 0, stream>>>(H1, H1, RED, L[0][7], L[0][8], NN * HD);

    // global_add_pool + MLP head -> d_out [64,64]
    k_zero<<<64, B, 0, stream>>>(POOL, NGR * HD);
    k_pool<<<NN, B, 0, stream>>>(H1, batch, POOL);
    k_mlp1<<<(NGR * (HD / 2) + 255) / 256, B, 0, stream>>>(POOL, mlp_w1, mlp_b1, TMP);
    k_mlp2<<<(NGR * NOUT + 255) / 256, B, 0, stream>>>(TMP, mlp_w2, mlp_b2,
                                                       (float*)d_out);
}
